// QinGNN_81432579932318
// MI455X (gfx1250) — matrix operand order repaired and checked by reading, not compiled, not run on hardware
//
#include <hip/hip_runtime.h>
#include <hip/hip_bf16.h>

typedef __attribute__((ext_vector_type(16))) _Float16 v16h;
typedef __attribute__((ext_vector_type(8)))  _Float16 v8h;
typedef __attribute__((ext_vector_type(4)))  _Float16 v4h;
typedef __attribute__((ext_vector_type(8)))  float    v8f;

// Block tile: 128(M) x 128(N) x 64(K); 256 threads = 8 waves, 4(M) x 2(N);
// each wave owns a 32x64 strip = 2x4 WMMA 16x16 tiles.
#define BM 128
#define BN 128
#define BK 64
#define LDS_A  (BK + 8)   // A tile row stride (halves): 72 -> 144B, 16B aligned
#define LDS_BT (BK + 8)   // transposed-B col stride (halves): 72 -> 144B

// ---------------------------------------------------------------------------
// Fragment loaders (wave32, v_wmma_f32_16x16x32_f16 layouts per ISA 7.12.2).
// Both are pure contiguous 16B LDS reads -> 2x ds_load_b128 each.
// ---------------------------------------------------------------------------
// A 16x32: lanes 0-15 row M=lane, K={0..7,16..23}; lanes 16-31 row M=lane-16,
// K={8..15,24..31}.
__device__ __forceinline__ v16h frag_a(const _Float16* __restrict__ p, int lane) {
    const _Float16* q = p + (lane & 15) * LDS_A + ((lane >> 4) << 3);
    v8h lo = *(const v8h*)(q);
    v8h hi = *(const v8h*)(q + 16);
    return __builtin_shufflevector(lo, hi, 0, 1, 2, 3, 4, 5, 6, 7,
                                   8, 9, 10, 11, 12, 13, 14, 15);
}

// B 32x16: lanes 0-15 col N=lane, K={0..7,16..23}; lanes 16-31 col N=lane-16, K={8..15,24..31}.
// Bt is stored transposed: Bt[col * LDS_BT + k], so this is contiguous in K.
__device__ __forceinline__ v16h frag_b(const _Float16* __restrict__ p, int lane) {
    const _Float16* q = p + (lane & 15) * LDS_BT + ((lane >> 4) << 3);
    v8h lo = *(const v8h*)(q);
    v8h hi = *(const v8h*)(q + 16);
    return __builtin_shufflevector(lo, hi, 0, 1, 2, 3, 4, 5, 6, 7,
                                   8, 9, 10, 11, 12, 13, 14, 15);
}

// Per-thread staging registers for one 128x64 A tile + 64x128 B tile.
struct TileRegs {
    v8h a[4];   // 4 x 8 contiguous halves of A
    v8h b[4];   // 4 K-rows x 8 N-cols sub-block of B
};

__device__ __forceinline__ void gload(TileRegs& r,
                                      const _Float16* __restrict__ Ag,
                                      const _Float16* __restrict__ Bg,
                                      int K, int N, int m0, int n0, int k0,
                                      int tid) {
#pragma unroll
    for (int i = 0; i < 4; ++i) {
        int off  = (tid + i * 256) * 8;
        int row  = off >> 6;          // / BK
        int colk = off & (BK - 1);
        r.a[i] = *(const v8h*)(Ag + (long long)(m0 + row) * K + (k0 + colk));
    }
    const int kb = (tid >> 4) * 4;    // 0..60
    const int nb = (tid & 15) * 8;    // 0..120
#pragma unroll
    for (int i = 0; i < 4; ++i)
        r.b[i] = *(const v8h*)(Bg + (long long)(k0 + kb + i) * N + (n0 + nb));
}

__device__ __forceinline__ void sstore(_Float16* __restrict__ As,
                                       _Float16* __restrict__ Bt,
                                       const TileRegs& r, int tid) {
#pragma unroll
    for (int i = 0; i < 4; ++i) {
        int off  = (tid + i * 256) * 8;
        int row  = off >> 6;
        int colk = off & (BK - 1);
        *(v8h*)(&As[row * LDS_A + colk]) = r.a[i];
    }
    const int kb = (tid >> 4) * 4;
    const int nb = (tid & 15) * 8;
    // transpose the 4(K) x 8(N) sub-block: one ds_store_b64 per column
#pragma unroll
    for (int i = 0; i < 8; ++i) {
        v4h t;
        t[0] = r.b[0][i]; t[1] = r.b[1][i]; t[2] = r.b[2][i]; t[3] = r.b[3][i];
        *(v4h*)(&Bt[(nb + i) * LDS_BT + kb]) = t;
    }
}

// ---------------------------------------------------------------------------
// Batched WMMA GEMM: C[b] = act(A[b] @ B[b] + bias), optional fused
// column-sum accumulation (global-average-pool numerator).
// Shapes are exact multiples of the tiles (M=1024, N=256, K in {128,256,1024}).
// ---------------------------------------------------------------------------
template <bool RELU, bool STORE, bool COLSUM>
__global__ void __launch_bounds__(256)
wmma_gemm_kernel(const _Float16* __restrict__ A, long long sA,
                 const _Float16* __restrict__ B, long long sB,
                 const float* __restrict__ bias,
                 _Float16* __restrict__ C, long long sC,
                 float* __restrict__ colsum,
                 int M, int N, int K) {
    __shared__ _Float16 As[2][BM * LDS_A];   // 2 x 18 KB
    __shared__ _Float16 Bt[2][BN * LDS_BT];  // 2 x 18 KB (transposed)

    const int bz = blockIdx.z;
    const _Float16* __restrict__ Ag = A + (long long)bz * sA;
    const _Float16* __restrict__ Bg = B + (long long)bz * sB;

    const int m0    = blockIdx.x * BM;
    const int n0    = blockIdx.y * BN;
    const int tid   = threadIdx.x;
    const int lane  = tid & 31;
    const int wid   = tid >> 5;
    const int waveM = wid >> 1;   // 0..3 -> 32-row strip
    const int waveN = wid & 1;    // 0..1 -> 64-col strip

    const v8f zacc = {};
    v8f acc[2][4];
#pragma unroll
    for (int tm = 0; tm < 2; ++tm)
#pragma unroll
        for (int tn = 0; tn < 4; ++tn) acc[tm][tn] = zacc;

    const int nsteps = K / BK;
    TileRegs regs;

    // pipeline preamble: tile 0 -> LDS buf0; tile 1 -> registers
    gload(regs, Ag, Bg, K, N, m0, n0, 0, tid);
    sstore(As[0], Bt[0], regs, tid);
    if (nsteps > 1) gload(regs, Ag, Bg, K, N, m0, n0, BK, tid);
    __syncthreads();

    for (int s = 0; s < nsteps; ++s) {
        const int cur = s & 1;
        // stage tile s+1 into the other buffer; issue global loads for s+2
        if (s + 1 < nsteps) sstore(As[cur ^ 1], Bt[cur ^ 1], regs, tid);
        if (s + 2 < nsteps) gload(regs, Ag, Bg, K, N, m0, n0, (s + 2) * BK, tid);

        const _Float16* __restrict__ as = As[cur];
        const _Float16* __restrict__ bs = Bt[cur];
#pragma unroll
        for (int kk = 0; kk < BK; kk += 32) {
            v16h af[2], bf[4];
#pragma unroll
            for (int tm = 0; tm < 2; ++tm)
                af[tm] = frag_a(&as[(waveM * 32 + tm * 16) * LDS_A + kk], lane);
#pragma unroll
            for (int tn = 0; tn < 4; ++tn)
                bf[tn] = frag_b(&bs[(waveN * 64 + tn * 16) * LDS_BT + kk], lane);
#pragma unroll
            for (int tm = 0; tm < 2; ++tm)
#pragma unroll
                for (int tn = 0; tn < 4; ++tn)
                    acc[tm][tn] = __builtin_amdgcn_wmma_f32_16x16x32_f16(
                        false, af[tm], false, bf[tn], (short)0, acc[tm][tn],
                        false, false);
        }
        __syncthreads();
    }

    // ----------------------- epilogue -----------------------
    // C/D layout: lane<16 -> col=lane, VGPR r -> M=r; lane>=16 -> col=lane-16,
    // VGPR r -> M=8+r.
    const int colLane = lane & 15;
    const int rowHalf = (lane >> 4) << 3;

    if (STORE) {
        _Float16* __restrict__ Cg = C + (long long)bz * sC;
#pragma unroll
        for (int tm = 0; tm < 2; ++tm) {
#pragma unroll
            for (int tn = 0; tn < 4; ++tn) {
                const int ncol = n0 + waveN * 64 + tn * 16 + colLane;
                const int rowb = m0 + waveM * 32 + tm * 16 + rowHalf;
                const float bv = bias ? bias[ncol] : 0.0f;
#pragma unroll
                for (int r = 0; r < 8; ++r) {
                    float v = acc[tm][tn][r] + bv;
                    if (RELU) v = fmaxf(v, 0.0f);
                    Cg[(long long)(rowb + r) * N + ncol] = (_Float16)v;
                }
            }
        }
    }
    if (COLSUM) {
        float* __restrict__ cs = colsum + (long long)bz * N;
#pragma unroll
        for (int tn = 0; tn < 4; ++tn) {
            const int ncol = n0 + waveN * 64 + tn * 16 + colLane;
            const float bv = bias ? bias[ncol] : 0.0f;
            float s = 0.0f;
#pragma unroll
            for (int tm = 0; tm < 2; ++tm)
#pragma unroll
                for (int r = 0; r < 8; ++r) {
                    float v = acc[tm][tn][r] + bv;
                    if (RELU) v = fmaxf(v, 0.0f);
                    s += v;
                }
            atomicAdd(&cs[ncol], s);
        }
    }
}

// ---------------------------------------------------------------------------
// fp32 -> f16 conversion (grid-stride)
// ---------------------------------------------------------------------------
__global__ void __launch_bounds__(256)
cvt_f32_f16_kernel(const float* __restrict__ in, _Float16* __restrict__ out,
                   long long n) {
    long long i = (long long)blockIdx.x * blockDim.x + threadIdx.x;
    long long stride = (long long)gridDim.x * blockDim.x;
    for (; i < n; i += stride) out[i] = (_Float16)in[i];
}

__global__ void __launch_bounds__(256)
zero_f32_kernel(float* __restrict__ p, int n) {
    int i = blockIdx.x * blockDim.x + threadIdx.x;
    if (i < n) p[i] = 0.0f;
}

// ---------------------------------------------------------------------------
// Head: g = colsum/N ; g@Wr1+br1 ; g@Wr2+br2 ; g@Wo+bo  -> out[b]
// ---------------------------------------------------------------------------
__global__ void __launch_bounds__(256)
head_kernel(const float* __restrict__ colsum,
            const float* __restrict__ Wr1, const float* __restrict__ br1,
            const float* __restrict__ Wr2, const float* __restrict__ br2,
            const float* __restrict__ Wo,  const float* __restrict__ bo,
            float* __restrict__ out, float inv_n) {
    __shared__ float s0[256];
    __shared__ float s1[256];
    __shared__ float red[256];
    const int b = blockIdx.x, t = threadIdx.x;

    s0[t] = colsum[b * 256 + t] * inv_n;
    __syncthreads();

    float acc = br1[t];
#pragma unroll 8
    for (int k = 0; k < 256; ++k) acc = fmaf(s0[k], Wr1[k * 256 + t], acc);
    s1[t] = acc;
    __syncthreads();

    acc = br2[t];
#pragma unroll 8
    for (int k = 0; k < 256; ++k) acc = fmaf(s1[k], Wr2[k * 256 + t], acc);
    red[t] = acc * Wo[t];
    __syncthreads();

#pragma unroll
    for (int s = 128; s > 0; s >>= 1) {
        if (t < s) red[t] += red[t + s];
        __syncthreads();
    }
    if (t == 0) out[b] = red[0] + bo[0];
}

// ---------------------------------------------------------------------------
// Driver
// ---------------------------------------------------------------------------
extern "C" void kernel_launch(void* const* d_in, const int* in_sizes, int n_in,
                              void* d_out, int out_size, void* d_ws, size_t ws_size,
                              hipStream_t stream) {
    constexpr long long B = 64, N = 1024, F = 128, H = 256;

    const float* x   = (const float*)d_in[0];
    const float* a   = (const float*)d_in[1];
    const float* W1  = (const float*)d_in[2];
    const float* b1  = (const float*)d_in[3];
    const float* W2  = (const float*)d_in[4];
    const float* b2  = (const float*)d_in[5];
    const float* Wr1 = (const float*)d_in[6];
    const float* br1 = (const float*)d_in[7];
    const float* Wr2 = (const float*)d_in[8];
    const float* br2 = (const float*)d_in[9];
    const float* Wo  = (const float*)d_in[10];
    const float* bo  = (const float*)d_in[11];
    float* out = (float*)d_out;

    char* ws = (char*)d_ws;
    size_t off = 0;
    auto alloc = [&](size_t bytes) -> void* {
        void* p = ws + off;
        off += (bytes + 255) & ~(size_t)255;
        return p;
    };

    _Float16* A_h  = (_Float16*)alloc(B * N * N * 2);   // 128 MB (fits in L2)
    _Float16* X_h  = (_Float16*)alloc(B * N * F * 2);   // 16 MB
    _Float16* W1_h = (_Float16*)alloc(F * H * 2);
    _Float16* W2_h = (_Float16*)alloc(H * H * 2);
    _Float16* XW   = (_Float16*)alloc(B * N * H * 2);   // 32 MB
    _Float16* H1   = (_Float16*)alloc(B * N * H * 2);   // 32 MB
    _Float16* HW   = (_Float16*)alloc(B * N * H * 2);   // 32 MB
    float*    cs   = (float*)alloc(B * H * 4);          // column sums

    // fp32 -> f16 staging
    cvt_f32_f16_kernel<<<8192, 256, 0, stream>>>(a,  A_h,  B * N * N);
    cvt_f32_f16_kernel<<<2048, 256, 0, stream>>>(x,  X_h,  B * N * F);
    cvt_f32_f16_kernel<<<32,   256, 0, stream>>>(W1, W1_h, F * H);
    cvt_f32_f16_kernel<<<64,   256, 0, stream>>>(W2, W2_h, H * H);
    zero_f32_kernel<<<(int)((B * H + 255) / 256), 256, 0, stream>>>(cs, (int)(B * H));

    const dim3 grid((unsigned)(N / BM), (unsigned)(H / BN), (unsigned)B);

    // 1) XW = X @ W1                         [1024x128]@[128x256]
    wmma_gemm_kernel<false, true, false><<<grid, 256, 0, stream>>>(
        X_h, N * F, W1_h, 0LL, nullptr, XW, N * H, nullptr,
        (int)N, (int)H, (int)F);

    // 2) H1 = relu(A @ XW + b1)              [1024x1024]@[1024x256]
    wmma_gemm_kernel<true, true, false><<<grid, 256, 0, stream>>>(
        A_h, N * N, XW, N * H, b1, H1, N * H, nullptr,
        (int)N, (int)H, (int)N);

    // 3) HW = H1 @ W2                        [1024x256]@[256x256]
    wmma_gemm_kernel<false, true, false><<<grid, 256, 0, stream>>>(
        H1, N * H, W2_h, 0LL, nullptr, HW, N * H, nullptr,
        (int)N, (int)H, (int)H);

    // 4) colsum += relu(A @ HW + b2)  (fused global-average-pool numerator)
    wmma_gemm_kernel<true, false, true><<<grid, 256, 0, stream>>>(
        A_h, N * N, HW, N * H, b2, nullptr, 0LL, cs,
        (int)N, (int)H, (int)N);

    // 5) head MLP -> out[64]
    head_kernel<<<(unsigned)B, 256, 0, stream>>>(
        cs, Wr1, br1, Wr2, br2, Wo, bo, out, 1.0f / (float)N);
}
